// Auto_Attn_15453292331527
// MI455X (gfx1250) — matrix operand order repaired and checked by reading, not compiled, not run on hardware
//
#include <hip/hip_runtime.h>
#include <hip/hip_bf16.h>

#define DEV static __device__ __forceinline__

constexpr int kB  = 4;
constexpr int kC  = 256;
constexpr int kC2 = 512;
constexpr int kCQ = 64;
constexpr int kN  = 4096;   // 64*64 pixels

typedef __attribute__((ext_vector_type(16))) __bf16 v16bf;
typedef __attribute__((ext_vector_type(8)))  __bf16 v8bf;
typedef __attribute__((ext_vector_type(8)))  float  v8f;

union AB16 { v16bf v; v8bf h[2]; unsigned short u[16]; };

// float -> bf16 round-to-nearest-even
DEV unsigned short f2bf(float f) {
  unsigned u = __builtin_bit_cast(unsigned, f);
  u += 0x7fffu + ((u >> 16) & 1u);
  return (unsigned short)(u >> 16);
}

// A-matrix 16x32 bf16 fragment: lane holds row M=lane%16.
// element e in [0,8)  -> K = (hi?8:0) + e      (contiguous 16B chunk)
// element e in [8,16) -> K = 16 + (hi?8:0) + (e-8) (second 16B chunk)
DEV v16bf load_afrag(const unsigned short* base /* row start + k0 */, int hi) {
  AB16 r;
  r.h[0] = *(const v8bf*)(base + hi * 8);
  r.h[1] = *(const v8bf*)(base + 16 + hi * 8);
  return r.v;
}

// B-matrix 32x16 bf16 fragment: lane holds column N=lane%16,
// same K order as A: e<8 -> K=(hi?8:0)+e, e>=8 -> K=16+(hi?8:0)+(e-8).
DEV v16bf load_bfrag(const unsigned short* col /* column start + k0 */, int hi) {
  AB16 r;
  r.h[0] = *(const v8bf*)(col + hi * 8);
  r.h[1] = *(const v8bf*)(col + 16 + hi * 8);
  return r.v;
}

DEV v16bf zfrag() {
  AB16 z;
  #pragma unroll
  for (int i = 0; i < 16; ++i) z.u[i] = 0;
  return z.v;
}

DEV v8f wmma_bf16(v16bf a, v16bf b, v8f c) {
  return __builtin_amdgcn_wmma_f32_16x16x32_bf16(false, a, false, b, (short)0, c,
                                                 false, false);
}

// Async copy of 16B/lane from global to LDS (gfx1250 GLOBAL_LOAD_ASYNC_TO_LDS_B128,
// GV addressing). Tracked by ASYNCcnt.
DEV void async_g2l_b128(unsigned lds_off, const void* gaddr) {
  asm volatile("global_load_async_to_lds_b128 %0, %1, off"
               :: "v"(lds_off), "v"((unsigned long long)(uintptr_t)gaddr)
               : "memory");
}
DEV void wait_async0() { asm volatile("s_wait_asynccnt 0x0" ::: "memory"); }

// ---------------------------------------------------------------------------
// Pack kernels
// ---------------------------------------------------------------------------

// vcol[b][c2][n] = bf16( c2<256 ? x : pre )   (channel-major, K-contig for PV B-frags)
__global__ void pack_vcol(const float* __restrict__ x, const float* __restrict__ pre,
                          unsigned short* __restrict__ vcol) {
  const size_t total = (size_t)kB * kC2 * kN;
  for (size_t i = (size_t)blockIdx.x * blockDim.x + threadIdx.x; i < total;
       i += (size_t)gridDim.x * blockDim.x) {
    size_t n = i % kN;
    size_t c2 = (i / kN) % kC2;
    size_t b = i / ((size_t)kN * kC2);
    float v = (c2 < (size_t)kC) ? x[(b * kC + c2) * kN + n]
                                : pre[(b * kC + (c2 - kC)) * kN + n];
    vcol[i] = f2bf(v);
  }
}

// xrow[b][n][c] = bf16(x[b][c][n])  (pixel-major transpose via LDS tiles)
__global__ __launch_bounds__(256) void pack_xrow(const float* __restrict__ x,
                                                 unsigned short* __restrict__ xrow) {
  __shared__ float t[32][33];
  int lx = threadIdx.x & 31, ly = threadIdx.x >> 5;       // 32 x 8
  int bt = blockIdx.x;
  int nt = bt & 127; bt >>= 7;                            // kN/32
  int ct = bt & 7;   bt >>= 3;                            // kC/32
  int b  = bt;
  int cbase = ct * 32, nbase = nt * 32;
  #pragma unroll
  for (int i = 0; i < 4; ++i)
    t[ly + i * 8][lx] = x[((size_t)b * kC + cbase + ly + i * 8) * kN + nbase + lx];
  __syncthreads();
  #pragma unroll
  for (int i = 0; i < 4; ++i)
    xrow[((size_t)b * kN + nbase + ly + i * 8) * kC + cbase + lx] =
        f2bf(t[lx][ly + i * 8]);
}

// plain fp32 -> bf16 cast (wq: [cq][c], ws: [co][ci] already K-contiguous)
__global__ void cast_bf(const float* __restrict__ w, unsigned short* __restrict__ wp,
                        size_t total) {
  for (size_t i = (size_t)blockIdx.x * blockDim.x + threadIdx.x; i < total;
       i += (size_t)gridDim.x * blockDim.x)
    wp[i] = f2bf(w[i]);
}

// OIHW (Co,Ci,3,3) -> [tap][co][ci] bf16 (ci contiguous for B-frags)
__global__ void pack_w3x3(const float* __restrict__ w, unsigned short* __restrict__ wp,
                          int Co, int Ci) {
  const size_t total = (size_t)Co * Ci * 9;
  for (size_t i = (size_t)blockIdx.x * blockDim.x + threadIdx.x; i < total;
       i += (size_t)gridDim.x * blockDim.x) {
    size_t ci = i % Ci;
    size_t co = (i / Ci) % Co;
    size_t tap = i / ((size_t)Ci * Co);
    wp[i] = f2bf(w[(co * Ci + ci) * 9 + tap]);
  }
}

// ---------------------------------------------------------------------------
// q = Wq x + bq : GEMM. Each wave: 32 pixels x 64 q-channels (K=256);
// weight B-frags reused across the two pixel tiles.
// ---------------------------------------------------------------------------
__global__ __launch_bounds__(256) void qgemm(const unsigned short* __restrict__ xrow,
                                             const unsigned short* __restrict__ wqp,
                                             const float* __restrict__ bq,
                                             unsigned short* __restrict__ qrow) {
  const int tid = threadIdx.x, wave = tid >> 5, lane = tid & 31;
  const int row = lane & 15, hi = lane >> 4;
  const int gw = blockIdx.x * 8 + wave;          // kB * (kN/32) waves
  const int b = gw >> 7, pt = gw & 127;
  const int m0 = pt * 32;
  const unsigned short* A0 = xrow + ((size_t)b * kN + m0 + row) * kC;
  const unsigned short* A1 = A0 + (size_t)16 * kC;
  v8f acc0[4] = {}, acc1[4] = {};
  for (int ks = 0; ks < 8; ++ks) {
    v16bf a0 = load_afrag(A0 + ks * 32, hi);
    v16bf a1 = load_afrag(A1 + ks * 32, hi);
    #pragma unroll
    for (int t = 0; t < 4; ++t) {
      v16bf bf = load_bfrag(wqp + (size_t)(t * 16 + row) * kC + ks * 32, hi);
      acc0[t] = wmma_bf16(a0, bf, acc0[t]);
      acc1[t] = wmma_bf16(a1, bf, acc1[t]);
    }
  }
  #pragma unroll
  for (int t = 0; t < 4; ++t) {
    float bias = bq[t * 16 + row];
    #pragma unroll
    for (int r = 0; r < 8; ++r) {
      int p = m0 + r + hi * 8;
      qrow[((size_t)b * kN + p) * kCQ + t * 16 + row] = f2bf(acc0[t][r] + bias);
      qrow[((size_t)b * kN + p + 16) * kCQ + t * 16 + row] = f2bf(acc1[t][r] + bias);
    }
  }
}

// ---------------------------------------------------------------------------
// Fused flash attention: S = Q^T K (Cq=64), online softmax, O = P * V
// V = concat(x, pre) (512 ch). K blocks (4KB, contiguous in qrow) are staged
// in LDS with double-buffered GLOBAL_LOAD_ASYNC_TO_LDS_B128, issued one
// iteration ahead; one barrier per iteration. Epilogue applies gamma/x
// residual and the alpha/mask blend, writing inp (raw + lrelu) bf16.
// Block: 8 waves; wave = (query tile of 16) x (128-channel quarter).
// ---------------------------------------------------------------------------
__global__ __launch_bounds__(256) void attn_fuse(
    const unsigned short* __restrict__ qrow, const unsigned short* __restrict__ vcol,
    const float* __restrict__ x, const float* __restrict__ pre,
    const float* __restrict__ mask, const float* __restrict__ gammap,
    const float* __restrict__ alphap, unsigned short* __restrict__ inp_raw,
    unsigned short* __restrict__ inp_lrelu) {
  __shared__ alignas(64) unsigned short Kbuf[2][32 * kCQ];   // 2 x 4KB key blocks
  __shared__ alignas(64) unsigned short Pbuf[8 * 16 * 32];
  const int tid = threadIdx.x, wave = tid >> 5, lane = tid & 31;
  const int row = lane & 15, hi = lane >> 4;
  const int b = blockIdx.x >> 7;               // kN/32 = 128 query blocks / batch
  const int qb = blockIdx.x & 127;
  const int qt = wave >> 2, cquarter = wave & 3;
  const int m0 = qb * 32 + qt * 16;
  const int c0 = cquarter * 128;
  unsigned short* Pw = &Pbuf[wave * 512];

  const unsigned short* kkeys = qrow + (size_t)b * kN * kCQ;  // keys, row-major
  const unsigned lds0 = (unsigned)(uintptr_t)&Kbuf[0][0] + tid * 16;
  const unsigned lds1 = (unsigned)(uintptr_t)&Kbuf[1][0] + tid * 16;

  // prologue: async-stage first key block
  async_g2l_b128(lds0, (const char*)kkeys + tid * 16);

  // Q A-fragments: 16 queries x 64 K
  const unsigned short* qA = qrow + ((size_t)b * kN + m0 + row) * kCQ;
  const v16bf qa0 = load_afrag(qA, hi);
  const v16bf qa1 = load_afrag(qA + 32, hi);

  v8f acc[8] = {};
  float rmax[8], rsum[8];
  #pragma unroll
  for (int r = 0; r < 8; ++r) { rmax[r] = -1e30f; rsum[r] = 0.0f; }

  const unsigned short* vbase = vcol + ((size_t)b * kC2 + c0 + row) * kN;

  for (int it = 0; it < kN / 32; ++it) {
    const int n0 = it * 32;
    const int cur = it & 1;
    wait_async0();          // this wave's async landed
    __syncthreads();        // -> every wave's async landed; prev reads done
    if (n0 + 32 < kN)       // stage next block into the other buffer
      async_g2l_b128(cur ? lds0 : lds1,
                     (const char*)(kkeys + (size_t)(n0 + 32) * kCQ) + tid * 16);

    // --- S = Q^T K : two 16x16 tiles over this 32-key block (from LDS) ---
    v8f s0 = {}, s1 = {};
    s0 = wmma_bf16(qa0, load_bfrag(&Kbuf[cur][(size_t)row * kCQ], hi), s0);
    s0 = wmma_bf16(qa1, load_bfrag(&Kbuf[cur][(size_t)row * kCQ + 32], hi), s0);
    s1 = wmma_bf16(qa0, load_bfrag(&Kbuf[cur][(size_t)(16 + row) * kCQ], hi), s1);
    s1 = wmma_bf16(qa1, load_bfrag(&Kbuf[cur][(size_t)(16 + row) * kCQ + 32], hi), s1);

    if (n0 + 32 < kN) __builtin_prefetch(vbase + n0 + 32, 0, 0);

    // --- online softmax (row stats live in the D-layout row mapping) ---
    float scale[8];
    #pragma unroll
    for (int r = 0; r < 8; ++r) {
      float m01 = fmaxf(s0[r], s1[r]);
      #pragma unroll
      for (int off = 1; off < 16; off <<= 1) m01 = fmaxf(m01, __shfl_xor(m01, off, 32));
      float nm = fmaxf(rmax[r], m01);
      scale[r] = __expf(rmax[r] - nm);
      rmax[r] = nm;
      float p0 = __expf(s0[r] - nm);
      float p1 = __expf(s1[r] - nm);
      float ps = p0 + p1;
      #pragma unroll
      for (int off = 1; off < 16; off <<= 1) ps += __shfl_xor(ps, off, 32);
      rsum[r] = rsum[r] * scale[r] + ps;
      const int m = r + hi * 8;                 // D-layout row
      Pw[m * 32 + row] = f2bf(p0);
      Pw[m * 32 + 16 + row] = f2bf(p1);
    }
    #pragma unroll
    for (int i = 0; i < 8; ++i)
      #pragma unroll
      for (int r = 0; r < 8; ++r) acc[i][r] *= scale[r];

    // P: LDS round-trip converts D-layout -> A-layout (wave-private)
    v16bf pa = load_afrag(Pw + row * 32, hi);

    // --- O += P * V (8 channel tiles of 16) ---
    const unsigned short* vb = vbase + n0;
    #pragma unroll
    for (int i = 0; i < 8; ++i)
      acc[i] = wmma_bf16(pa, load_bfrag(vb + (size_t)i * 16 * kN, hi), acc[i]);
  }

  // --- epilogue: normalize, residual/blend, write inp raw + lrelu ---
  const float gamma = gammap[0], alpha = alphap[0];
  float inv[8];
  #pragma unroll
  for (int r = 0; r < 8; ++r) inv[r] = 1.0f / rsum[r];
  #pragma unroll
  for (int i = 0; i < 8; ++i) {
    const int c = c0 + i * 16 + row;
    #pragma unroll
    for (int r = 0; r < 8; ++r) {
      const int p = m0 + r + hi * 8;
      const float a = acc[i][r] * inv[r];
      float outv;
      if (c < kC) {                                         // wave-uniform branch
        outv = gamma * a + x[((size_t)b * kC + c) * kN + p];
      } else {
        float pv = pre[((size_t)b * kC + (c - kC)) * kN + p];
        float Mv = mask[(size_t)b * 3 * kN + p];
        outv = alpha * Mv * a + (1.0f - Mv) * pv;
      }
      const size_t o = ((size_t)b * kN + p) * kC2 + c;
      inp_raw[o] = f2bf(outv);
      inp_lrelu[o] = f2bf(outv >= 0.0f ? outv : 0.1f * outv);
    }
  }
}

// ---------------------------------------------------------------------------
// conv1: 3x3, 512 -> 256, input pre-lrelu'd, SAME pad; output lrelu'd bf16.
// Implicit GEMM: each wave = 32 pixels x 16 out-channels; weight B-frags
// shared by the two pixel tiles.
// ---------------------------------------------------------------------------
__global__ __launch_bounds__(256) void conv3x3_a(
    const unsigned short* __restrict__ act,   // inp_lrelu [B][N][512]
    const unsigned short* __restrict__ wp,    // [9][256][512]
    const float* __restrict__ bias, unsigned short* __restrict__ outb) {
  const int tid = threadIdx.x, wave = tid >> 5, lane = tid & 31;
  const int row = lane & 15, hi = lane >> 4;
  const int gw = blockIdx.x * 8 + wave;        // kB*(kN/32)*16 waves
  const int ct = gw & 15, pt = (gw >> 4) & 127, b = gw >> 11;
  const int y = pt >> 1, x0 = (pt & 1) * 32;
  const int co0 = ct * 16;
  v8f acc0 = {}, acc1 = {};
  #pragma unroll
  for (int tap = 0; tap < 9; ++tap) {
    const int dy = tap / 3 - 1, dx = tap % 3 - 1;
    const int yy = y + dy;
    if ((unsigned)yy >= 64u) continue;                      // wave-uniform skip
    const int xa = x0 + row + dx, xb = x0 + 16 + row + dx;
    const bool va = (unsigned)xa < 64u, vb = (unsigned)xb < 64u;
    const unsigned short* Aa = act + ((size_t)b * kN + yy * 64 + xa) * kC2;
    const unsigned short* Ab = act + ((size_t)b * kN + yy * 64 + xb) * kC2;
    const unsigned short* Wb = wp + ((size_t)tap * kC + co0 + row) * kC2;
    for (int ks = 0; ks < 16; ++ks) {
      v16bf bf = load_bfrag(Wb + ks * 32, hi);
      acc0 = wmma_bf16(va ? load_afrag(Aa + ks * 32, hi) : zfrag(), bf, acc0);
      acc1 = wmma_bf16(vb ? load_afrag(Ab + ks * 32, hi) : zfrag(), bf, acc1);
    }
  }
  const float bv = bias[co0 + row];
  #pragma unroll
  for (int r = 0; r < 8; ++r) {
    float h0 = acc0[r] + bv, h1 = acc1[r] + bv;
    h0 = h0 >= 0.0f ? h0 : 0.1f * h0;                       // lrelu for conv2 input
    h1 = h1 >= 0.0f ? h1 : 0.1f * h1;
    const size_t p = (size_t)b * kN + y * 64 + x0 + r + hi * 8;
    outb[p * kC + co0 + row] = f2bf(h0);
    outb[(p + 16) * kC + co0 + row] = f2bf(h1);
  }
}

// ---------------------------------------------------------------------------
// conv2 (3x3, 256->256) + 1x1 shortcut (512->256) fused; fp32 NCHW out.
// Same 32-pixel x 16-channel wave tiling with weight reuse.
// ---------------------------------------------------------------------------
__global__ __launch_bounds__(256) void conv3x3_b(
    const unsigned short* __restrict__ act,      // lrelu(h1) [B][N][256]
    const unsigned short* __restrict__ w2p,      // [9][256][256]
    const unsigned short* __restrict__ inp_raw,  // [B][N][512]
    const unsigned short* __restrict__ wsp,      // [256][512]
    const float* __restrict__ b2, const float* __restrict__ bs,
    float* __restrict__ out) {
  const int tid = threadIdx.x, wave = tid >> 5, lane = tid & 31;
  const int row = lane & 15, hi = lane >> 4;
  const int gw = blockIdx.x * 8 + wave;
  const int ct = gw & 15, pt = (gw >> 4) & 127, b = gw >> 11;
  const int y = pt >> 1, x0 = (pt & 1) * 32;
  const int co0 = ct * 16;
  v8f acc0 = {}, acc1 = {};
  #pragma unroll
  for (int tap = 0; tap < 9; ++tap) {
    const int dy = tap / 3 - 1, dx = tap % 3 - 1;
    const int yy = y + dy;
    if ((unsigned)yy >= 64u) continue;
    const int xa = x0 + row + dx, xb = x0 + 16 + row + dx;
    const bool va = (unsigned)xa < 64u, vb = (unsigned)xb < 64u;
    const unsigned short* Aa = act + ((size_t)b * kN + yy * 64 + xa) * kC;
    const unsigned short* Ab = act + ((size_t)b * kN + yy * 64 + xb) * kC;
    const unsigned short* Wb = w2p + ((size_t)tap * kC + co0 + row) * kC;
    for (int ks = 0; ks < 8; ++ks) {
      v16bf bf = load_bfrag(Wb + ks * 32, hi);
      acc0 = wmma_bf16(va ? load_afrag(Aa + ks * 32, hi) : zfrag(), bf, acc0);
      acc1 = wmma_bf16(vb ? load_afrag(Ab + ks * 32, hi) : zfrag(), bf, acc1);
    }
  }
  // shortcut 1x1 over raw inp (K=512), no edge masking needed
  {
    const unsigned short* Aa = inp_raw + ((size_t)b * kN + y * 64 + x0 + row) * kC2;
    const unsigned short* Ab = Aa + (size_t)16 * kC2;
    const unsigned short* Wb = wsp + (size_t)(co0 + row) * kC2;
    for (int ks = 0; ks < 16; ++ks) {
      v16bf bf = load_bfrag(Wb + ks * 32, hi);
      acc0 = wmma_bf16(load_afrag(Aa + ks * 32, hi), bf, acc0);
      acc1 = wmma_bf16(load_afrag(Ab + ks * 32, hi), bf, acc1);
    }
  }
  const float bv = b2[co0 + row] + bs[co0 + row];
  #pragma unroll
  for (int r = 0; r < 8; ++r) {
    const size_t o = ((size_t)b * kC + co0 + row) * kN + y * 64 + x0 + r + hi * 8;
    out[o] = acc0[r] + bv;
    out[o + 16] = acc1[r] + bv;
  }
}

// ---------------------------------------------------------------------------

extern "C" void kernel_launch(void* const* d_in, const int* in_sizes, int n_in,
                              void* d_out, int out_size, void* d_ws, size_t ws_size,
                              hipStream_t stream) {
  (void)in_sizes; (void)n_in; (void)out_size; (void)ws_size;
  const float* x     = (const float*)d_in[0];
  const float* pre   = (const float*)d_in[1];
  const float* mask  = (const float*)d_in[2];
  const float* wq    = (const float*)d_in[3];
  const float* bq    = (const float*)d_in[4];
  const float* gamma = (const float*)d_in[5];
  const float* alpha = (const float*)d_in[6];
  const float* w1    = (const float*)d_in[7];
  const float* b1    = (const float*)d_in[8];
  const float* w2    = (const float*)d_in[9];
  const float* b2    = (const float*)d_in[10];
  const float* ws    = (const float*)d_in[11];
  const float* bs    = (const float*)d_in[12];
  float* out = (float*)d_out;

  char* wsp_base = (char*)d_ws;
  size_t off = 0;
  auto alloc = [&](size_t bytes) {
    void* p = wsp_base + off;
    off += (bytes + 255) & ~(size_t)255;
    return p;
  };
  unsigned short* qrow      = (unsigned short*)alloc((size_t)kB * kN * kCQ * 2);
  unsigned short* vcol      = (unsigned short*)alloc((size_t)kB * kC2 * kN * 2);
  unsigned short* xrow      = (unsigned short*)alloc((size_t)kB * kN * kC * 2);
  unsigned short* inp_raw   = (unsigned short*)alloc((size_t)kB * kN * kC2 * 2);
  unsigned short* inp_lrelu = (unsigned short*)alloc((size_t)kB * kN * kC2 * 2);
  unsigned short* h1row     = (unsigned short*)alloc((size_t)kB * kN * kC * 2);
  unsigned short* wqp       = (unsigned short*)alloc((size_t)kCQ * kC * 2);
  unsigned short* w1p       = (unsigned short*)alloc((size_t)9 * kC * kC2 * 2);
  unsigned short* w2p       = (unsigned short*)alloc((size_t)9 * kC * kC * 2);
  unsigned short* wspk      = (unsigned short*)alloc((size_t)kC * kC2 * 2);

  // 1) pack / layout transforms
  pack_vcol<<<4096, 256, 0, stream>>>(x, pre, vcol);
  pack_xrow<<<4096, 256, 0, stream>>>(x, xrow);
  cast_bf<<<64, 256, 0, stream>>>(wq, wqp, (size_t)kCQ * kC);
  pack_w3x3<<<1024, 256, 0, stream>>>(w1, w1p, kC, kC2);
  pack_w3x3<<<1024, 256, 0, stream>>>(w2, w2p, kC, kC);
  cast_bf<<<512, 256, 0, stream>>>(ws, wspk, (size_t)kC * kC2);

  // 2) q projection (WMMA GEMM, 32 pixels/wave)
  qgemm<<<kB * (kN / 32) / 8, 256, 0, stream>>>(xrow, wqp, bq, qrow);

  // 3) fused flash attention (async K staging) + residual/mask epilogue
  attn_fuse<<<kB * (kN / 32), 256, 0, stream>>>(qrow, vcol, x, pre, mask, gamma,
                                                alpha, inp_raw, inp_lrelu);

  // 4) conv1 (WMMA implicit GEMM, 32 pixels/wave)
  conv3x3_a<<<kB * (kN / 32) * (kC / 16) / 8, 256, 0, stream>>>(inp_lrelu, w1p, b1,
                                                                h1row);

  // 5) conv2 + shortcut (WMMA implicit GEMM) -> fp32 NCHW output
  conv3x3_b<<<kB * (kN / 32) * (kC / 16) / 8, 256, 0, stream>>>(h1row, w2p, inp_raw,
                                                                wspk, b2, bs, out);
}
